// Decoder_56667798503957
// MI455X (gfx1250) — matrix operand order repaired and checked by reading, not compiled, not run on hardware
//
#include <hip/hip_runtime.h>
#include <cstdint>
#include <cstddef>

// Problem dims (match reference)
#define BB   256   // batch
#define TT   64    // time steps
#define DD   256   // hidden
#define LLAT 256   // latent
#define VV   8192  // vocab
#define HH   1024  // mlp hidden
#define NLAYER 2
#define TDIM (3*DD)

typedef __attribute__((ext_vector_type(16))) __bf16 v16bf;
typedef __attribute__((ext_vector_type(8)))  __bf16 v8bf;
typedef __attribute__((ext_vector_type(8)))  float  v8f;

static __device__ __forceinline__ __bf16 f2bf(float f) {
  unsigned u = __builtin_bit_cast(unsigned, f);
  u = (u + 0x7FFFu + ((u >> 16) & 1u)) >> 16;   // round-to-nearest-even
  unsigned short s = (unsigned short)u;
  return __builtin_bit_cast(__bf16, s);
}

static __device__ __forceinline__ v16bf combine16(v8bf lo, v8bf hi) {
  v16bf r;
#pragma unroll
  for (int i = 0; i < 8; ++i) { r[i] = lo[i]; r[8 + i] = hi[i]; }
  return r;
}

// ---------- one-time prep kernels ----------
__global__ void k_cvt_bf16(const float* __restrict__ src, __bf16* __restrict__ dst, int n) {
  int i = blockIdx.x * blockDim.x + threadIdx.x;
  if (i < n) dst[i] = f2bf(src[i]);
}

// src f32 [R,C] -> dst bf16 [C,R]
__global__ void k_transpose_bf16(const float* __restrict__ src, __bf16* __restrict__ dst,
                                 int R, int C) {
  int i = blockIdx.x * blockDim.x + threadIdx.x;
  if (i < R * C) {
    int r = i / C, c = i - r * C;
    dst[c * R + r] = f2bf(src[i]);
  }
}

__global__ void k_init(const float* __restrict__ hidden,
                       float* __restrict__ h0, float* __restrict__ h1,
                       __bf16* __restrict__ h0b, __bf16* __restrict__ h1b,
                       __bf16* __restrict__ msgb, float* __restrict__ loss_acc) {
  int i = blockIdx.x * blockDim.x + threadIdx.x;
  if (i < BB * DD) {
    float v = hidden[i];
    __bf16 vb = f2bf(v);
    h0[i] = v; h1[i] = v;
    h0b[i] = vb; h1b[i] = vb;
    msgb[i] = f2bf(0.0f);
  }
  if (i < BB) loss_acc[i] = 0.0f;
}

// ---------- WMMA GEMM: C[M,N] = act(A_bf16[M,K] * W_bf16[N,K]^T + bias) ----------
// One wave computes a 32x64 output tile: 2 A fragments x 4 B fragments ->
// 8 accumulators, each B fragment reused for 2 WMMAs (halves B traffic/FLOP).
// 4 waves per block. M%32==0, N%64==0, K%32==0 required.
template <int RELU, int OUT_BF16>
__global__ __launch_bounds__(128)
void k_gemm(const __bf16* __restrict__ A, const __bf16* __restrict__ W,
            const float* __restrict__ bias, void* __restrict__ Cout,
            int M, int N, int K) {
  const int lane = threadIdx.x & 31;
  const int wave = threadIdx.x >> 5;
  const int ntiles = N >> 6;               // 64-wide tiles
  const int mtiles = M >> 5;               // 32-row tiles
  const int tile = blockIdx.x * 4 + wave;
  if (tile >= mtiles * ntiles) return;
  const int mt = tile / ntiles;
  const int m0 = mt << 5;
  const int n0 = (tile - mt * ntiles) << 6;

  const int half = lane >> 4;   // 0 or 1
  const int l15  = lane & 15;

  // A fragments per lane: one 16-element K run per lane (matches B)
  const __bf16* ap0 = A + (size_t)(m0 +      l15) * K + half * 16;
  const __bf16* ap1 = A + (size_t)(m0 + 16 + l15) * K + half * 16;
  // B fragments: one 16-element K run per lane (contiguous k = 16*half + s, same as A), W is [N,K]
  const __bf16* wp0 = W + (size_t)(n0 +  0 + l15) * K + half * 16;
  const __bf16* wp1 = W + (size_t)(n0 + 16 + l15) * K + half * 16;
  const __bf16* wp2 = W + (size_t)(n0 + 32 + l15) * K + half * 16;
  const __bf16* wp3 = W + (size_t)(n0 + 48 + l15) * K + half * 16;

  v8f acc00 = {}, acc01 = {}, acc02 = {}, acc03 = {};
  v8f acc10 = {}, acc11 = {}, acc12 = {}, acc13 = {};

#pragma unroll 2
  for (int k0 = 0; k0 < K; k0 += 32) {
    // prefetch weight streams ahead (global_prefetch_b8)
    __builtin_prefetch(wp0 + k0 + 256);
    __builtin_prefetch(wp1 + k0 + 256);
    __builtin_prefetch(wp2 + k0 + 256);
    __builtin_prefetch(wp3 + k0 + 256);

    v16bf af0 = combine16(*(const v8bf*)(ap0 + k0), *(const v8bf*)(ap0 + k0 + 8));
    v16bf af1 = combine16(*(const v8bf*)(ap1 + k0), *(const v8bf*)(ap1 + k0 + 8));

    v16bf wf0 = combine16(*(const v8bf*)(wp0 + k0), *(const v8bf*)(wp0 + k0 + 8));
    acc00 = __builtin_amdgcn_wmma_f32_16x16x32_bf16(false, af0, false, wf0, (short)0, acc00, false, false);
    acc10 = __builtin_amdgcn_wmma_f32_16x16x32_bf16(false, af1, false, wf0, (short)0, acc10, false, false);

    v16bf wf1 = combine16(*(const v8bf*)(wp1 + k0), *(const v8bf*)(wp1 + k0 + 8));
    acc01 = __builtin_amdgcn_wmma_f32_16x16x32_bf16(false, af0, false, wf1, (short)0, acc01, false, false);
    acc11 = __builtin_amdgcn_wmma_f32_16x16x32_bf16(false, af1, false, wf1, (short)0, acc11, false, false);

    v16bf wf2 = combine16(*(const v8bf*)(wp2 + k0), *(const v8bf*)(wp2 + k0 + 8));
    acc02 = __builtin_amdgcn_wmma_f32_16x16x32_bf16(false, af0, false, wf2, (short)0, acc02, false, false);
    acc12 = __builtin_amdgcn_wmma_f32_16x16x32_bf16(false, af1, false, wf2, (short)0, acc12, false, false);

    v16bf wf3 = combine16(*(const v8bf*)(wp3 + k0), *(const v8bf*)(wp3 + k0 + 8));
    acc03 = __builtin_amdgcn_wmma_f32_16x16x32_bf16(false, af0, false, wf3, (short)0, acc03, false, false);
    acc13 = __builtin_amdgcn_wmma_f32_16x16x32_bf16(false, af1, false, wf3, (short)0, acc13, false, false);
  }

  // C/D layout: VGPR i -> row base + half*8 + i, col base + (lane&15)
  const int r0 = m0 + half * 8;
  const int r1 = m0 + 16 + half * 8;
  auto store_tile = [&](const v8f& acc, int crow, int ccol) {
    const float bv = bias ? bias[ccol] : 0.0f;
#pragma unroll
    for (int i = 0; i < 8; ++i) {
      float v = acc[i] + bv;
      if (RELU) v = v > 0.0f ? v : 0.0f;
      if (OUT_BF16) ((__bf16*)Cout)[(size_t)(crow + i) * N + ccol] = f2bf(v);
      else          ((float*)Cout)[(size_t)(crow + i) * N + ccol] = v;
    }
  };
  store_tile(acc00, r0, n0 +  0 + l15);
  store_tile(acc10, r1, n0 +  0 + l15);
  store_tile(acc01, r0, n0 + 16 + l15);
  store_tile(acc11, r1, n0 + 16 + l15);
  store_tile(acc02, r0, n0 + 32 + l15);
  store_tile(acc12, r1, n0 + 32 + l15);
  store_tile(acc03, r0, n0 + 48 + l15);
  store_tile(acc13, r1, n0 + 48 + l15);
}

// ---------- GRU gate pointwise: h (f32) updated, bf16 copies for next GEMMs ----------
__global__ __launch_bounds__(256)
void k_gru_pointwise(const float* __restrict__ gi, const float* __restrict__ gh,
                     float* __restrict__ h, __bf16* __restrict__ hbf,
                     __bf16* __restrict__ xbf) {
  int i = blockIdx.x * blockDim.x + threadIdx.x;   // i < BB*DD
  int b = i >> 8;          // DD == 256
  int d = i & 255;
  const float* gib = gi + (size_t)b * TDIM;
  const float* ghb = gh + (size_t)b * TDIM;
  float i_r = gib[d], i_z = gib[DD + d], i_n = gib[2 * DD + d];
  float h_r = ghb[d], h_z = ghb[DD + d], h_n = ghb[2 * DD + d];
  float r  = 1.0f / (1.0f + __expf(-(i_r + h_r)));
  float zg = 1.0f / (1.0f + __expf(-(i_z + h_z)));
  float n  = tanhf(i_n + r * h_n);
  float hv = (1.0f - zg) * n + zg * h[i];
  h[i]   = hv;
  __bf16 hb = f2bf(hv);
  hbf[i] = hb;
  xbf[i] = hb;
}

// ---------- per-row logits reduction: argmax + logsumexp + CE + bf16 gather ----------
__global__ __launch_bounds__(256)
void k_row_reduce(const float* __restrict__ dot, const int* __restrict__ answers,
                  const int* __restrict__ lengths, const __bf16* __restrict__ monoB,
                  int t, __bf16* __restrict__ msgb, float* __restrict__ loss_acc,
                  float* __restrict__ dec_out) {
  const int b = blockIdx.x;
  const int tid = threadIdx.x;
  const float* row = dot + (size_t)b * VV;
  __shared__ float sm[256];
  __shared__ int   sa[256];
  __shared__ int   s_target;

  // pass 1: max + first-occurrence argmax
  float m = -3.402823466e38f; int am = 0;
  for (int v = tid; v < VV; v += 256) {
    float x = row[v];
    if (x > m) { m = x; am = v; }
  }
  sm[tid] = m; sa[tid] = am;
  __syncthreads();
  for (int s = 128; s > 0; s >>= 1) {
    if (tid < s) {
      float m2 = sm[tid + s]; int a2 = sa[tid + s];
      if (m2 > sm[tid] || (m2 == sm[tid] && a2 < sa[tid])) { sm[tid] = m2; sa[tid] = a2; }
    }
    __syncthreads();
  }
  const float gmax = sm[0];
  const int   garg = sa[0];
  __syncthreads();

  // pass 2: sum exp
  float s = 0.0f;
  for (int v = tid; v < VV; v += 256) s += __expf(row[v] - gmax);
  sm[tid] = s;
  __syncthreads();
  for (int st = 128; st > 0; st >>= 1) {
    if (tid < st) sm[tid] += sm[tid + st];
    __syncthreads();
  }

  if (tid == 0) {
    int len = lengths[b];
    int idx = (t < len - 1) ? t : (len - 1);        // min(step, len-1)
    int target = answers[b * TT + idx];
    s_target = target;
    float lse = gmax + __logf(sm[0]);
    float ce  = lse - row[target];
    float mask = (len > t) ? 1.0f : 0.0f;
    loss_acc[b] += ce * mask;                       // one block owns row b -> deterministic
    dec_out[(size_t)t * BB + b] = (float)garg;
  }
  __syncthreads();
  // teacher forcing: next message (bf16) = monomers[target]
  msgb[(size_t)b * DD + tid] = monoB[(size_t)s_target * DD + tid];
}

__global__ __launch_bounds__(256)
void k_finalize(const float* __restrict__ loss_acc, float* __restrict__ out) {
  __shared__ float sm[256];
  int tid = threadIdx.x;
  sm[tid] = loss_acc[tid];
  __syncthreads();
  for (int s = 128; s > 0; s >>= 1) {
    if (tid < s) sm[tid] += sm[tid + s];
    __syncthreads();
  }
  if (tid == 0) out[0] = sm[0] / (float)BB;
}

// ---------- host side ----------
extern "C" void kernel_launch(void* const* d_in, const int* in_sizes, int n_in,
                              void* d_out, int out_size, void* d_ws, size_t ws_size,
                              hipStream_t stream) {
  const float* z       = (const float*)d_in[0];   // [B,L]
  const float* mono    = (const float*)d_in[1];   // [V,D]
  const int*   answers = (const int*)  d_in[2];   // [B,T]
  const int*   lengths = (const int*)  d_in[3];   // [B]
  const float* Wz      = (const float*)d_in[4];   // [L,D]
  const float* bz      = (const float*)d_in[5];   // [D]
  const float* Wih     = (const float*)d_in[6];   // [NL,3D,D]
  const float* Whh     = (const float*)d_in[7];   // [NL,3D,D]
  const float* bih     = (const float*)d_in[8];   // [NL,3D]
  const float* bhh     = (const float*)d_in[9];   // [NL,3D]
  const float* W1      = (const float*)d_in[10];  // [D,H]
  const float* b1      = (const float*)d_in[11];  // [H]
  const float* W2      = (const float*)d_in[12];  // [H,D]
  const float* b2      = (const float*)d_in[13];  // [D]
  float* out = (float*)d_out;

  // workspace carve-up
  char* p = (char*)d_ws;
  auto alloc = [&](size_t bytes) -> void* {
    void* q = (void*)p;
    p += (bytes + 255) & ~(size_t)255;
    return q;
  };
  // bf16 weights (B-operands, [N,K] row-major)
  __bf16* WzT    = (__bf16*)alloc((size_t)DD * LLAT * 2);       // [D,L]
  __bf16* WihB   = (__bf16*)alloc((size_t)NLAYER * TDIM * DD * 2);
  __bf16* WhhB   = (__bf16*)alloc((size_t)NLAYER * TDIM * DD * 2);
  __bf16* W1T    = (__bf16*)alloc((size_t)HH * DD * 2);         // [H,D]
  __bf16* W2T    = (__bf16*)alloc((size_t)DD * HH * 2);         // [D,H]
  __bf16* monoB  = (__bf16*)alloc((size_t)VV * DD * 2);         // [V,D]
  __bf16* zB     = (__bf16*)alloc((size_t)BB * LLAT * 2);
  // f32 state
  float* hidden  = (float*)alloc((size_t)BB * DD * 4);
  float* h0      = (float*)alloc((size_t)BB * DD * 4);
  float* h1      = (float*)alloc((size_t)BB * DD * 4);
  float* gi      = (float*)alloc((size_t)BB * TDIM * 4);
  float* gh      = (float*)alloc((size_t)BB * TDIM * 4);
  float* dotbuf  = (float*)alloc((size_t)BB * VV * 4);
  float* lossacc = (float*)alloc((size_t)BB * 4);
  // bf16 activations (A-operands)
  __bf16* h0b    = (__bf16*)alloc((size_t)BB * DD * 2);
  __bf16* h1b    = (__bf16*)alloc((size_t)BB * DD * 2);
  __bf16* msgb   = (__bf16*)alloc((size_t)BB * DD * 2);
  __bf16* x1b    = (__bf16*)alloc((size_t)BB * DD * 2);
  __bf16* x2b    = (__bf16*)alloc((size_t)BB * DD * 2);
  __bf16* hid1b  = (__bf16*)alloc((size_t)BB * HH * 2);
  __bf16* mapb   = (__bf16*)alloc((size_t)BB * DD * 2);
  (void)ws_size; (void)in_sizes; (void)n_in; (void)out_size;

  auto cvt = [&](const float* s, __bf16* d, int n) {
    k_cvt_bf16<<<(n + 255) / 256, 256, 0, stream>>>(s, d, n);
  };
  auto tpose = [&](const float* s, __bf16* d, int R, int C) {
    k_transpose_bf16<<<(R * C + 255) / 256, 256, 0, stream>>>(s, d, R, C);
  };
  // GEMM variants: f32-out, bf16-out, bf16-out + relu
  auto gemm_f32 = [&](const __bf16* A, const __bf16* W, const float* bias, float* C,
                      int M, int N, int K) {
    int blocks = ((M / 32) * (N / 64) + 3) / 4;
    k_gemm<0, 0><<<blocks, 128, 0, stream>>>(A, W, bias, (void*)C, M, N, K);
  };
  auto gemm_bf = [&](const __bf16* A, const __bf16* W, const float* bias, __bf16* C,
                     int M, int N, int K, bool relu) {
    int blocks = ((M / 32) * (N / 64) + 3) / 4;
    if (relu) k_gemm<1, 1><<<blocks, 128, 0, stream>>>(A, W, bias, (void*)C, M, N, K);
    else      k_gemm<0, 1><<<blocks, 128, 0, stream>>>(A, W, bias, (void*)C, M, N, K);
  };

  // ---- one-time prep: bf16 weights + bf16 z ----
  tpose(Wz, WzT, LLAT, DD);                         // Wz [L,D] -> [D,L]
  cvt(Wih, WihB, NLAYER * TDIM * DD);               // already [3D, D] per layer
  cvt(Whh, WhhB, NLAYER * TDIM * DD);
  tpose(W1, W1T, DD, HH);                           // W1 [D,H] -> [H,D]
  tpose(W2, W2T, HH, DD);                           // W2 [H,D] -> [D,H]
  cvt(mono, monoB, VV * DD);                        // monomers [V,D] serves as B^T directly
  cvt(z, zB, BB * LLAT);

  // hidden = z @ Wz + bz ; h0 = h1 = hidden ; msg = 0 ; loss = 0
  gemm_f32(zB, WzT, bz, hidden, BB, DD, LLAT);
  k_init<<<(BB * DD + 255) / 256, 256, 0, stream>>>(hidden, h0, h1, h0b, h1b, msgb, lossacc);

  const int gp_blocks = (BB * DD + 255) / 256;
  for (int t = 0; t < TT; ++t) {
    // GRU layer 0 (x = msg)
    gemm_f32(msgb, WihB,               bih,        gi, BB, TDIM, DD);
    gemm_f32(h0b,  WhhB,               bhh,        gh, BB, TDIM, DD);
    k_gru_pointwise<<<gp_blocks, 256, 0, stream>>>(gi, gh, h0, h0b, x1b);
    // GRU layer 1 (x = h0_new)
    gemm_f32(x1b, WihB + (size_t)TDIM * DD, bih + TDIM, gi, BB, TDIM, DD);
    gemm_f32(h1b, WhhB + (size_t)TDIM * DD, bhh + TDIM, gh, BB, TDIM, DD);
    k_gru_pointwise<<<gp_blocks, 256, 0, stream>>>(gi, gh, h1, h1b, x2b);
    // MLP (bf16 chained outputs, fused bias/relu)
    gemm_bf(x2b,   W1T, b1, hid1b, BB, HH, DD, true);
    gemm_bf(hid1b, W2T, b2, mapb,  BB, DD, HH, false);
    // logits (f32 out)
    gemm_f32(mapb, monoB, nullptr, dotbuf, BB, VV, DD);
    // argmax / logsumexp / CE / teacher forcing
    k_row_reduce<<<BB, 256, 0, stream>>>(dotbuf, answers, lengths, monoB,
                                         t, msgb, lossacc, out + 1);
  }
  k_finalize<<<1, 256, 0, stream>>>(lossacc, out);
}
